// CNN_51015621542651
// MI455X (gfx1250) — matrix operand order repaired and checked by reading, not compiled, not run on hardware
//
#include <hip/hip_runtime.h>
#include <hip/hip_bf16.h>
#include <math.h>

typedef __attribute__((ext_vector_type(16))) _Float16 v16h;
typedef __attribute__((ext_vector_type(8)))  _Float16 v8h;
typedef __attribute__((ext_vector_type(8)))  float    v8f;

#define N_IMG 16

#if __has_builtin(__builtin_amdgcn_global_load_async_to_lds_b64)
#define ASYNC_B 1
typedef int v2i __attribute__((vector_size(2 * sizeof(int))));
#define GAS64(p) ((__attribute__((address_space(1))) v2i*)(v2i*)(void*)(p))
#define LAS64(p) ((__attribute__((address_space(3))) v2i*)(v2i*)(void*)(p))
#else
#define ASYNC_B 0
#endif

// 16-byte A-fragment piece with explicit (possibly sub-16B) alignment
template<int A> struct H8T;
template<> struct __attribute__((packed, aligned(8))) H8T<8> { v8h v; };
template<> struct __attribute__((packed, aligned(2))) H8T<2> { v8h v; };

// ---------------------------------------------------------------------------
// zero-fill f16 buffer, 8 halfs (16B) per thread
// ---------------------------------------------------------------------------
__global__ void zero16(_Float16* __restrict__ p, int n8) {
  int i = blockIdx.x * blockDim.x + threadIdx.x;
  if (i < n8) *(v8h*)(p + (size_t)i * 8) = (v8h){};
}

// ---------------------------------------------------------------------------
// input f32 [16,1,64^3] -> padded f16 [16, 70,70,70] (interior at +3)
// ---------------------------------------------------------------------------
__global__ void cvt_pad0(const float* __restrict__ in, _Float16* __restrict__ out, int n) {
  int i = blockIdx.x * blockDim.x + threadIdx.x;
  if (i >= n) return;
  const int img = i >> 18;            // 64^3 = 262144
  const int sp  = i & 262143;
  const int d = sp >> 12, h = (sp >> 6) & 63, w = sp & 63;
  out[(size_t)img * 343000 + ((size_t)(d + 3) * 70 + (h + 3)) * 70 + (w + 3)] = (_Float16)in[i];
}

// ---------------------------------------------------------------------------
// Weight synthesis + WMMA B-layout packing.
// K ordering: rows r=(kd*5+kh) of length ROWLEN (5*CIN padded to mult of 8);
// within a row, o = kw*CIN + ci (matches channel-last activation memory).
// B element: t = ((nt*KC + chunk)*32 + lane)*16 + e ;
//   n = (lane&15)+16*nt ; K-slot q = 16*(lane>=16)+e ; k' = chunk*32+q.
// ---------------------------------------------------------------------------
__device__ inline void pack_layer(_Float16* dst, const float* __restrict__ W,
                                  int CIN, int COUT, int KC, int ROWLEN,
                                  const float (*Bb)[125], int tid, int nthreads) {
  const int total = 2 * KC * 512;
  const int VALID = CIN * 5;
  for (int t = tid; t < total; t += nthreads) {
    int e    = t & 15;
    int lane = (t >> 4) & 31;
    int c    = (t >> 9) % KC;
    int nt   = t / (KC * 512);
    int n    = (lane & 15) + nt * 16;
    int kp   = c * 32 + ((lane >> 4) << 4) + e;   // global K' index
    int r    = kp / ROWLEN;
    int o    = kp - r * ROWLEN;
    float val = 0.0f;
    if (n < COUT && r < 25 && o < VALID) {
      int kw = o / CIN;
      int ci = o - kw * CIN;
      int kd = r / 5, kh = r - kd * 5;
      int v  = kd * 25 + kh * 5 + kw;
      const float* w = W + (n * CIN + ci) * 3;
      val = w[0] * Bb[0][v] + w[1] * Bb[1][v] + w[2] * Bb[2][v];
    }
    dst[t] = (_Float16)val;
  }
}

__global__ void synth_weights(const float* __restrict__ W0, const float* __restrict__ W1,
                              const float* __restrict__ W2,
                              _Float16* K0p, _Float16* K1p, _Float16* K2p) {
  __shared__ float Bb[3][125];
  __shared__ float nrm[3];
  const int tid = threadIdx.x;
  for (int t = tid; t < 375; t += blockDim.x) {
    int j = t / 125, v = t % 125;
    int kd = v / 25, kh = (v / 5) % 5, kw = v % 5;
    float dx = kd - 2.f, dy = kh - 2.f, dz = kw - 2.f;
    float r  = sqrtf(dx * dx + dy * dy + dz * dz);
    float d  = (r - (float)j) * (1.0f / 0.6f);
    Bb[j][v] = expf(-0.5f * d * d);
  }
  __syncthreads();
  if (tid < 3) {
    float s = 0.f;
    for (int v = 0; v < 125; ++v) s += Bb[tid][v] * Bb[tid][v];
    nrm[tid] = rsqrtf(s);
  }
  __syncthreads();
  for (int t = tid; t < 375; t += blockDim.x) Bb[t / 125][t % 125] *= nrm[t / 125];
  __syncthreads();
  pack_layer(K0p, W0,  1, 23,  7,   8, Bb, tid, blockDim.x);
  pack_layer(K1p, W1, 20, 23, 82, 104, Bb, tid, blockDim.x);
  pack_layer(K2p, W2, 20, 20, 82, 104, Bb, tid, blockDim.x);
}

// ---------------------------------------------------------------------------
// Implicit-GEMM conv (stride 2, pad 3, 5^3) on padded channel-last f16 input.
// One wave: MT M-tiles x 32 padded Couts -> 2*MT WMMA per K-chunk.
// A fragment per lane per chunk = two contiguous 16-byte global loads.
// B per chunk (2KB, shared by whole block) is double-buffered in LDS via
// GLOBAL_LOAD_ASYNC_TO_LDS (ASYNCcnt) + s_wait_asynccnt + split barrier.
// ---------------------------------------------------------------------------
template<int CIN, int COUT, int SOUT, int SP, int KC, int ROWLEN, int MT>
__global__ __launch_bounds__(256)
void conv_wmma(const _Float16* __restrict__ act,
               const _Float16* __restrict__ wpack,
               float* __restrict__ y) {
  using H8 = H8T<(CIN % 4 == 0) ? 8 : 2>;
  const int S3     = SOUT * SOUT * SOUT;
  const int MTILES = (N_IMG * S3) / 16;
  const int wid    = blockIdx.x * (blockDim.x >> 5) + (threadIdx.x >> 5);
  const int tid    = threadIdx.x;

  // per-block K-octet offset table (row offset + intra-row offset)
  __shared__ int otab[KC * 4];
#if ASYNC_B
  __shared__ __align__(32) _Float16 bsh[2][1024];   // [buf][ntile*512 + ...]
#endif
  for (int t = tid; t < KC * 4; t += blockDim.x) {
    int c  = t >> 2;
    int hi = (t >> 1) & 1;
    int j  = t & 1;
    int O  = c * 4 + hi * 2 + j;      // octet index, k' = O*8
    int k8 = O * 8;
    int r  = k8 / ROWLEN;
    int o  = k8 - r * ROWLEN;
    otab[t] = (r < 25) ? (((r / 5) * SP + (r % 5)) * SP * CIN + o) : 0;
  }
  __syncthreads();

  const int lane = tid & 31;
  const int hi   = lane >> 4;

  // per-tile A-row base pointers (clamped for tiles past the end; stores masked)
  const _Float16* base[MT];
  #pragma unroll
  for (int t = 0; t < MT; ++t) {
    int tb = wid * MT + t;
    int mt = tb < MTILES ? tb : MTILES - 1;
    int m  = mt * 16 + (lane & 15);
    int img = m / S3;
    int sp  = m - img * S3;
    int od  = sp / (SOUT * SOUT);
    sp     -= od * SOUT * SOUT;
    int oh  = sp / SOUT;
    int ow  = sp - oh * SOUT;
    size_t voxbase = (((size_t)(od * 2) * SP + (oh * 2)) * SP + (ow * 2)) * CIN;
    base[t] = act + (size_t)img * SP * SP * SP * CIN + voxbase;
  }

  v8f acc0[MT];
  v8f acc1[MT];
  #pragma unroll
  for (int t = 0; t < MT; ++t) { acc0[t] = (v8f){}; acc1[t] = (v8f){}; }

#if ASYNC_B
  // cooperative async stage of chunk c's B (2 n-tiles x 1KB) into buffer p
  auto stage = [&](int c, int p) {
    const int nt = tid >> 7;            // 0..1
    const int q  = tid & 127;           // 8-byte segment within the 1KB n-tile
    const _Float16* src = wpack + (size_t)(nt * KC + c) * 512 + q * 4;
    __builtin_amdgcn_global_load_async_to_lds_b64(
        GAS64(src), LAS64(&bsh[p][nt * 512 + q * 4]), 0, 0);
  };
  stage(0, 0);
  asm volatile("s_wait_asynccnt 0x0" ::: "memory");
  __syncthreads();
#else
  const _Float16* wp0 = wpack + (size_t)lane * 16;               // n-tile 0
  const _Float16* wp1 = wpack + ((size_t)(KC * 32) + lane) * 16; // n-tile 1
#endif

  #pragma unroll 1
  for (int c = 0; c < KC; ++c) {
    const int2 oo = *(const int2*)(otab + ((c << 2) + (hi << 1)));
    v16h a[MT];
    #pragma unroll
    for (int t = 0; t < MT; ++t) {
      H8 lo = *(const H8*)(base[t] + oo.x);   // K-slots hi*16 + 0..7
      H8 hv = *(const H8*)(base[t] + oo.y);   // K-slots hi*16 + 8..15
      a[t] = __builtin_shufflevector(lo.v, hv.v,
                                     0, 1, 2, 3, 4, 5, 6, 7,
                                     8, 9, 10, 11, 12, 13, 14, 15);
    }
#if ASYNC_B
    if (c + 1 < KC) stage(c + 1, (c + 1) & 1);            // overlap with WMMA
    const v16h b0 = *(const v16h*)&bsh[c & 1][lane * 16];
    const v16h b1 = *(const v16h*)&bsh[c & 1][512 + lane * 16];
#else
    const v16h b0 = *(const v16h*)(wp0 + (size_t)c * 512);
    const v16h b1 = *(const v16h*)(wp1 + (size_t)c * 512);
    int cn = (c + 1 < KC) ? (c + 1) : c;
    __builtin_prefetch(wp0 + (size_t)cn * 512, 0, 3);
    __builtin_prefetch(wp1 + (size_t)cn * 512, 0, 3);
#endif
    #pragma unroll
    for (int t = 0; t < MT; ++t) {
      acc0[t] = __builtin_amdgcn_wmma_f32_16x16x32_f16(false, a[t], false, b0, (short)0, acc0[t], false, false);
      acc1[t] = __builtin_amdgcn_wmma_f32_16x16x32_f16(false, a[t], false, b1, (short)0, acc1[t], false, false);
    }
#if ASYNC_B
    if (c + 1 < KC) {                                     // uniform over block
      asm volatile("s_wait_asynccnt 0x0" ::: "memory");
      __syncthreads();
    }
#endif
  }

  // D layout: lanes 0-15 VGPR r -> M=r ; lanes 16-31 -> M=r+8 ; N = lane&15
  const int n0 = lane & 15;
  #pragma unroll
  for (int t = 0; t < MT; ++t) {
    const int tb = wid * MT + t;
    if (tb >= MTILES) break;          // uniform per wave
    #pragma unroll
    for (int r = 0; r < 8; ++r) {
      const int mm  = tb * 16 + hi * 8 + r;
      const int im2 = mm / S3;
      const int sp2 = mm - im2 * S3;
      y[((size_t)im2 * COUT + n0) * S3 + sp2] = acc0[t][r];
      if (n0 + 16 < COUT)
        y[((size_t)im2 * COUT + (n0 + 16)) * S3 + sp2] = acc1[t][r];
    }
  }
}

// ---------------------------------------------------------------------------
// Gated capsule mixing -> next layer's padded channel-last f16 activations.
// out = [relu(y[0:5]), y[5:8]*sig(y[20]), y[8:13]*sig(y[21]), y[13:20]*sig(y[22])]
// ---------------------------------------------------------------------------
template<int SOUT, int SP>
__global__ void gate_kernel(const float* __restrict__ y, _Float16* __restrict__ act, int total) {
  const int S3 = SOUT * SOUT * SOUT;
  int v = blockIdx.x * blockDim.x + threadIdx.x;
  if (v >= total) return;
  const int img = v / S3;
  int sp = v - img * S3;
  const int d = sp / (SOUT * SOUT);
  sp -= d * SOUT * SOUT;
  const int h = sp / SOUT;
  const int w = sp - h * SOUT;

  const float* yb = y + ((size_t)img * 23) * S3 + ((size_t)d * SOUT + h) * SOUT + w;
  float ch[23];
  #pragma unroll
  for (int c = 0; c < 23; ++c) ch[c] = yb[(size_t)c * S3];
  const float g0 = 1.f / (1.f + expf(-ch[20]));
  const float g1 = 1.f / (1.f + expf(-ch[21]));
  const float g2 = 1.f / (1.f + expf(-ch[22]));
  _Float16 o[20];
  #pragma unroll
  for (int c = 0; c < 5; ++c)   o[c] = (_Float16)fmaxf(ch[c], 0.f);
  #pragma unroll
  for (int c = 5; c < 8; ++c)   o[c] = (_Float16)(ch[c] * g0);
  #pragma unroll
  for (int c = 8; c < 13; ++c)  o[c] = (_Float16)(ch[c] * g1);
  #pragma unroll
  for (int c = 13; c < 20; ++c) o[c] = (_Float16)(ch[c] * g2);

  _Float16* ob = (_Float16*)__builtin_assume_aligned(
      act + ((size_t)img * SP * SP * SP + (((size_t)(d + 3) * SP) + (h + 3)) * SP + (w + 3)) * 20, 8);
  #pragma unroll
  for (int c = 0; c < 20; ++c) ob[c] = o[c];
}

// ---------------------------------------------------------------------------
// AvgSpacial + FC1(relu) + FC2 — single block, f32
// ---------------------------------------------------------------------------
__global__ void head_kernel(const float* __restrict__ y2,
                            const float* __restrict__ fc1w, const float* __restrict__ fc1b,
                            const float* __restrict__ fc2w, const float* __restrict__ fc2b,
                            float* __restrict__ out) {
  __shared__ float mean[N_IMG][20];
  __shared__ float hbuf[N_IMG][50];
  const int tid = threadIdx.x;
  for (int t = tid; t < N_IMG * 20; t += blockDim.x) {
    int img = t / 20, c = t % 20;
    const float* p = y2 + ((size_t)img * 20 + c) * 1000;
    float s = 0.f;
    for (int v = 0; v < 1000; ++v) s += p[v];
    mean[img][c] = s * 1e-3f;
  }
  __syncthreads();
  for (int t = tid; t < N_IMG * 50; t += blockDim.x) {
    int img = t / 50, j = t % 50;
    float s = fc1b[j];
    for (int c = 0; c < 20; ++c) s += mean[img][c] * fc1w[j * 20 + c];
    hbuf[img][j] = fmaxf(s, 0.f);
  }
  __syncthreads();
  for (int t = tid; t < N_IMG * 2; t += blockDim.x) {
    int img = t / 2, oc = t % 2;
    float s = fc2b[oc];
    for (int j = 0; j < 50; ++j) s += hbuf[img][j] * fc2w[oc * 50 + j];
    out[img * 2 + oc] = s;
  }
}

// ---------------------------------------------------------------------------
extern "C" void kernel_launch(void* const* d_in, const int* in_sizes, int n_in,
                              void* d_out, int out_size, void* d_ws, size_t ws_size,
                              hipStream_t stream) {
  const float* inp  = (const float*)d_in[0];
  const float* W0   = (const float*)d_in[1];
  const float* W1   = (const float*)d_in[2];
  const float* W2   = (const float*)d_in[3];
  const float* fc1w = (const float*)d_in[4];
  const float* fc1b = (const float*)d_in[5];
  const float* fc2w = (const float*)d_in[6];
  const float* fc2b = (const float*)d_in[7];
  float* out = (float*)d_out;
  (void)in_sizes; (void)n_in; (void)out_size; (void)ws_size;

  char* ws = (char*)d_ws;
  size_t off = 0;
  auto take = [&](size_t bytes) -> char* {
    char* p = ws + off;
    off += (bytes + 255) & ~(size_t)255;
    return p;
  };

  _Float16* pad0 = (_Float16*)take(5488000ull  * 2); // [16,70^3] f16 (CIN=1)
  _Float16* K0p  = (_Float16*)take(7168ull     * 2);
  _Float16* K1p  = (_Float16*)take(83968ull    * 2);
  _Float16* K2p  = (_Float16*)take(83968ull    * 2);
  float*    y0   = (float*)   take(13224816ull * 4); // [16,23,33^3]
  _Float16* pad1 = (_Float16*)take(20480000ull * 2); // [16,40^3,20] channel-last
  float*    y1   = (float*)   take(2146176ull  * 4); // [16,23,18^3]
  _Float16* pad2 = (_Float16*)take(4423680ull  * 2); // [16,24^3,20] channel-last
  float*    y2   = (float*)   take(320000ull   * 4); // [16,20,10^3]
  (void)take(4096);                                  // tail slack for octet overreads

  // zero padded activation buffers (deterministic each call)
  zero16<<<(686000  + 255) / 256, 256, 0, stream>>>(pad0, 686000);
  zero16<<<(2560000 + 255) / 256, 256, 0, stream>>>(pad1, 2560000);
  zero16<<<(552960  + 255) / 256, 256, 0, stream>>>(pad2, 552960);
  // input -> padded f16
  cvt_pad0<<<(4194304 + 255) / 256, 256, 0, stream>>>(inp, pad0, 4194304);
  // synthesize + pack conv weights
  synth_weights<<<1, 256, 0, stream>>>(W0, W1, W2, K0p, K1p, K2p);

  // layer 0: 35937 M-tiles / MT=4 -> 8985 waves
  conv_wmma<1, 23, 33, 70, 7, 8, 4><<<(8985 + 7) / 8, 256, 0, stream>>>(pad0, K0p, y0);
  gate_kernel<33, 40><<<(574992 + 255) / 256, 256, 0, stream>>>(y0, pad1, 574992);
  // layer 1 (dominant): 5832 M-tiles / MT=2 -> 2916 waves, 82 K-chunks
  conv_wmma<20, 23, 18, 40, 82, 104, 2><<<(2916 + 7) / 8, 256, 0, stream>>>(pad1, K1p, y1);
  gate_kernel<18, 24><<<(93312 + 255) / 256, 256, 0, stream>>>(y1, pad2, 93312);
  // layer 2: 1000 M-tiles / MT=2 -> 500 waves
  conv_wmma<20, 20, 10, 24, 82, 104, 2><<<(500 + 7) / 8, 256, 0, stream>>>(pad2, K2p, y2);
  // mean + FCs
  head_kernel<<<1, 256, 0, stream>>>(y2, fc1w, fc1b, fc2w, fc2b, out);
}
